// AttentionLayer_79216376807461
// MI455X (gfx1250) — matrix operand order repaired and checked by reading, not compiled, not run on hardware
//
#include <hip/hip_runtime.h>
#include <hip/hip_bf16.h>

#define DEVI __device__ __forceinline__

typedef __bf16 bf16;
typedef bf16  v16bf __attribute__((ext_vector_type(16)));
typedef bf16  v8bf  __attribute__((ext_vector_type(8)));
typedef float v8f   __attribute__((ext_vector_type(8)));
typedef unsigned int u32x4 __attribute__((ext_vector_type(4)));
typedef int i32x8 __attribute__((ext_vector_type(8)));
typedef int i32x4 __attribute__((ext_vector_type(4)));

constexpr int B_  = 8;
constexpr int CIN = 512;
constexpr int CO  = 512;
constexpr int CK  = 256;
constexpr int N_  = 2048;
constexpr int CALL = CK + CK + CO; // 1024 stacked output rows (q|k|v)

// ---- workspace layout (bytes, 256-aligned) ----
constexpr size_t OFF_WALL = 0;
constexpr size_t SZ_WALL  = (size_t)CALL * CIN * 2;
constexpr size_t OFF_XT   = OFF_WALL + SZ_WALL;                    // bf16 [B][N][CIN]
constexpr size_t SZ_XT    = (size_t)B_ * N_ * CIN * 2;
constexpr size_t OFF_QT   = OFF_XT + SZ_XT;                        // bf16 [B][N][CK]
constexpr size_t SZ_QT    = (size_t)B_ * N_ * CK * 2;
constexpr size_t OFF_KT   = OFF_QT + SZ_QT;                        // bf16 [B][N][CK]
constexpr size_t OFF_V    = OFF_KT + SZ_QT;                        // bf16 [B][CO][N]
constexpr size_t SZ_V     = (size_t)B_ * CO * N_ * 2;
constexpr size_t OFF_M    = OFF_V + SZ_V;                          // f32 [B][N]
constexpr size_t SZ_M     = (size_t)B_ * N_ * 4;
constexpr size_t OFF_S    = OFF_M + SZ_M;                          // f32 [B][N]
constexpr size_t OFF_PT   = OFF_S + SZ_M;                          // bf16 [B][N][N]

// ---- WMMA fragment helpers (gfx1250 wave32, 16-bit 16x16x32) ----
DEVI v16bf load_a_frag(const bf16* __restrict__ row, int k0, int half) {
    v8bf lo = *reinterpret_cast<const v8bf*>(row + k0 + half * 8);
    v8bf hi = *reinterpret_cast<const v8bf*>(row + k0 + 16 + half * 8);
    return __builtin_shufflevector(lo, hi, 0,1,2,3,4,5,6,7,8,9,10,11,12,13,14,15);
}
DEVI v16bf load_a_frag_lds(const bf16* row, int k0, int half) {
    v8bf lo = *reinterpret_cast<const v8bf*>(row + k0 + half * 8);
    v8bf hi = *reinterpret_cast<const v8bf*>(row + k0 + 16 + half * 8);
    return __builtin_shufflevector(lo, hi, 0,1,2,3,4,5,6,7,8,9,10,11,12,13,14,15);
}
DEVI v16bf load_b_frag(const bf16* __restrict__ btrow, int k0, int half) {
    v8bf lo = *reinterpret_cast<const v8bf*>(btrow + k0 + half * 8);
    v8bf hi = *reinterpret_cast<const v8bf*>(btrow + k0 + 16 + half * 8);
    return __builtin_shufflevector(lo, hi, 0,1,2,3,4,5,6,7,8,9,10,11,12,13,14,15);
}
DEVI v8f wmma_bf16(v16bf a, v16bf b, v8f c) {
    return __builtin_amdgcn_wmma_f32_16x16x32_bf16(false, a, false, b, (short)0, c, false, false);
}
DEVI v8f vzero() { v8f z = {0.f,0.f,0.f,0.f,0.f,0.f,0.f,0.f}; return z; }

// ---- TDM: 2-D tile (rows x cols, bf16) from global tensor into LDS ----
// tensor: row-major [tensor_dim1 x tensor_dim0] bf16, tile = [tile_rows x tile_cols]
// starting at tile_start.
DEVI void tdm_load_2d_bf16(unsigned lds_off, const bf16* tile_start,
                           int tensor_dim0, int tensor_dim1,
                           int tile_cols, int tile_rows, int row_stride) {
    unsigned long long ga = (unsigned long long)(uintptr_t)tile_start;
    u32x4 g0;
    g0[0] = 1u;                                               // count=1 (valid user D#)
    g0[1] = lds_off;                                          // LDS byte address
    g0[2] = (unsigned)(ga & 0xFFFFFFFFu);                     // global_addr[31:0]
    g0[3] = (unsigned)((ga >> 32) & 0x01FFFFFFu) | (2u << 30);// addr[56:32] | type=2
    i32x8 g1;
    g1[0] = (1 << 16);                                        // data_size=1 -> 2 bytes
    g1[1] = (int)(((unsigned)tensor_dim0 & 0xFFFFu) << 16);   // tensor_dim0 lo16 @ [63:48]
    g1[2] = (int)((((unsigned)tensor_dim0 >> 16) & 0xFFFFu)   // tensor_dim0 hi16 @ [79:64]
                | (((unsigned)tensor_dim1 & 0xFFFFu) << 16)); // tensor_dim1 lo16 @ [95:80]
    g1[3] = (int)((((unsigned)tensor_dim1 >> 16) & 0xFFFFu)   // tensor_dim1 hi16 @ [111:96]
                | (((unsigned)tile_cols) << 16));             // tile_dim0 @ [127:112]
    g1[4] = tile_rows;                                        // tile_dim1 @ [143:128], tile_dim2=0
    g1[5] = row_stride;                                       // tensor_dim0_stride[31:0]
    g1[6] = 0;
    g1[7] = 0;
    i32x4 z4 = {0, 0, 0, 0};
    i32x8 z8 = {0, 0, 0, 0, 0, 0, 0, 0};
    __builtin_amdgcn_tensor_load_to_lds(g0, g1, z4, z4, z8, 0);
}

// ---- kernel 1: pack Wq|Wk|Wv -> bf16 Wall[1024][512] ----
__global__ void k_pack_w(const float* __restrict__ Wq, const float* __restrict__ Wk,
                         const float* __restrict__ Wv, bf16* __restrict__ Wall) {
    int idx = blockIdx.x * 256 + threadIdx.x;
    int o = idx >> 9, c = idx & (CIN - 1);
    float f = (o < CK) ? Wq[o * CIN + c]
            : (o < 2 * CK) ? Wk[(o - CK) * CIN + c]
            : Wv[(o - 2 * CK) * CIN + c];
    Wall[idx] = (bf16)f;
}

// ---- kernel 2: x f32 [b][c][n] -> xT bf16 [b][n][c] (LDS-tiled transpose) ----
__global__ void k_xpose(const float* __restrict__ x, bf16* __restrict__ xT) {
    __shared__ float t[32][33];
    int n0 = blockIdx.x * 32, c0 = blockIdx.y * 32, b = blockIdx.z;
    int tx = threadIdx.x, ty = threadIdx.y;              // 32 x 8
    const float* xb = x + (size_t)b * CIN * N_;
#pragma unroll
    for (int i = 0; i < 4; ++i)
        t[ty + 8 * i][tx] = xb[(size_t)(c0 + ty + 8 * i) * N_ + n0 + tx];
    __syncthreads();
    bf16* xTb = xT + (size_t)b * N_ * CIN;
#pragma unroll
    for (int i = 0; i < 4; ++i)
        xTb[(size_t)(n0 + ty + 8 * i) * CIN + c0 + tx] = (bf16)t[tx][ty + 8 * i];
}

// ---- kernel 3: projections y = Wall * x_b ; write qT,kT (transposed), v (natural) ----
__global__ void k_proj(const bf16* __restrict__ Wall, const bf16* __restrict__ xT,
                       bf16* __restrict__ qT, bf16* __restrict__ kT, bf16* __restrict__ v) {
    int wave = threadIdx.x >> 5, lane = threadIdx.x & 31;
    int half = lane >> 4, lr = lane & 15;
    int bid = blockIdx.x;
    int b  = bid / (64 * 8);
    int r0 = bid % (64 * 8);
    int ot = r0 >> 3, ng = r0 & 7;
    int o0 = ot * 16;
    int n0 = ng * 256 + wave * 64;

    const bf16* arow = Wall + (size_t)(o0 + lr) * CIN;
    const bf16* xTb  = xT + (size_t)b * N_ * CIN;

    v8f acc[4];
#pragma unroll
    for (int t = 0; t < 4; ++t) acc[t] = vzero();

    for (int k0 = 0; k0 < CIN; k0 += 32) {
        v16bf a = load_a_frag(arow, k0, half);
#pragma unroll
        for (int t = 0; t < 4; ++t) {
            v16bf bb = load_b_frag(xTb + (size_t)(n0 + t * 16 + lr) * CIN, k0, half);
            acc[t] = wmma_bf16(a, bb, acc[t]);
        }
    }
    if (o0 < 2 * CK) {
        bf16* dst = (o0 < CK) ? qT : kT;
        int ob = (o0 < CK) ? o0 : (o0 - CK);
#pragma unroll
        for (int t = 0; t < 4; ++t) {
            int n = n0 + t * 16 + lr;
            v8bf pk;
#pragma unroll
            for (int r = 0; r < 8; ++r) pk[r] = (bf16)acc[t][r];
            *reinterpret_cast<v8bf*>(dst + ((size_t)b * N_ + n) * CK + ob + half * 8) = pk;
        }
    } else {
        int ob = o0 - 2 * CK;
#pragma unroll
        for (int t = 0; t < 4; ++t) {
            int n = n0 + t * 16 + lr;
#pragma unroll
            for (int r = 0; r < 8; ++r)
                v[((size_t)b * CO + ob + half * 8 + r) * N_ + n] = (bf16)acc[t][r];
        }
    }
}

// ---- kernel 4: row stats of sim = kT * q : online max / sum-exp over j ----
__global__ void k_rowstats(const bf16* __restrict__ kT, const bf16* __restrict__ qT,
                           float* __restrict__ Mrow, float* __restrict__ Srow) {
    int wave = threadIdx.x >> 5, lane = threadIdx.x & 31;
    int half = lane >> 4, lr = lane & 15;
    int w = blockIdx.x * 4 + wave;
    int b = w >> 7, it = w & 127;
    int i0 = it * 16;

    const bf16* kTrow = kT + ((size_t)b * N_ + i0 + lr) * CK;
    v16bf a[8];
#pragma unroll
    for (int kk = 0; kk < 8; ++kk) a[kk] = load_a_frag(kTrow, kk * 32, half);

    float mrun[8], srun[8];
#pragma unroll
    for (int r = 0; r < 8; ++r) { mrun[r] = -__builtin_inff(); srun[r] = 0.f; }

    const bf16* qTb = qT + (size_t)b * N_ * CK;
    for (int j0 = 0; j0 < N_; j0 += 16) {
        const bf16* brow = qTb + (size_t)(j0 + lr) * CK;
        if (j0 + 64 < N_)
            __builtin_prefetch(qTb + (size_t)(j0 + 64 + lr) * CK, 0, 0);
        v8f acc = vzero();
#pragma unroll
        for (int kk = 0; kk < 8; ++kk)
            acc = wmma_bf16(a[kk], load_b_frag(brow, kk * 32, half), acc);
#pragma unroll
        for (int r = 0; r < 8; ++r) {
            float xv = acc[r];
            float mn = fmaxf(mrun[r], xv);
            srun[r] = srun[r] * __expf(mrun[r] - mn) + __expf(xv - mn);
            mrun[r] = mn;
        }
    }
#pragma unroll
    for (int r = 0; r < 8; ++r) {
        float m = mrun[r], s = srun[r];
#pragma unroll
        for (int mask = 1; mask <= 8; mask <<= 1) {
            float mo = __shfl_xor(m, mask, 32);
            float so = __shfl_xor(s, mask, 32);
            float mn = fmaxf(m, mo);
            s = s * __expf(m - mn) + so * __expf(mo - mn);
            m = mn;
        }
        mrun[r] = m; srun[r] = s;
    }
    if (lr == 0) {
#pragma unroll
        for (int r = 0; r < 8; ++r) {
            int i = i0 + half * 8 + r;
            Mrow[b * N_ + i] = mrun[r];
            Srow[b * N_ + i] = srun[r];
        }
    }
}

// ---- kernel 5: Pt[b][j][i] = exp(sim[i][j]-M[i])/S[i] (bf16, A-layout for final GEMM) ----
__global__ void k_pt(const bf16* __restrict__ kT, const bf16* __restrict__ qT,
                     const float* __restrict__ Mrow, const float* __restrict__ Srow,
                     bf16* __restrict__ Pt) {
    int wave = threadIdx.x >> 5, lane = threadIdx.x & 31;
    int half = lane >> 4, lr = lane & 15;
    int w = blockIdx.x * 4 + wave;
    int b  = w / (128 * 16);
    int rr = w % (128 * 16);
    int it = rr >> 4, jg = rr & 15;
    int i0 = it * 16;

    const bf16* kTrow = kT + ((size_t)b * N_ + i0 + lr) * CK;
    v16bf a[8];
#pragma unroll
    for (int kk = 0; kk < 8; ++kk) a[kk] = load_a_frag(kTrow, kk * 32, half);

    float mv[8], rs[8];
#pragma unroll
    for (int r = 0; r < 8; ++r) {
        int i = i0 + half * 8 + r;
        mv[r] = Mrow[b * N_ + i];
        rs[r] = 1.0f / Srow[b * N_ + i];
    }
    const bf16* qTb = qT + (size_t)b * N_ * CK;
    for (int jt = 0; jt < 8; ++jt) {
        int j = jg * 128 + jt * 16 + lr;
        v8f acc = vzero();
        const bf16* brow = qTb + (size_t)j * CK;
#pragma unroll
        for (int kk = 0; kk < 8; ++kk)
            acc = wmma_bf16(a[kk], load_b_frag(brow, kk * 32, half), acc);
        v8bf pk;
#pragma unroll
        for (int r = 0; r < 8; ++r)
            pk[r] = (bf16)(__expf(acc[r] - mv[r]) * rs[r]);
        *reinterpret_cast<v8bf*>(Pt + ((size_t)b * N_ + j) * N_ + i0 + half * 8) = pk;
    }
}

// ---- kernel 6: out[b][m][c] = sum_k Pt[b][m][k] * v[b][c][k] ----
// Block = 4 waves sharing one 16-row A strip of Pt. TDM double-buffers 16x256
// A chunks into LDS (TENSORcnt-tracked) while waves compute; A frags via ds_load.
constexpr int KCH = 256;                    // K chunk
constexpr int NCH = N_ / KCH;               // 8 chunks
__global__ void k_out(const bf16* __restrict__ Pt, const bf16* __restrict__ v,
                      float* __restrict__ out) {
    __shared__ bf16 abuf[2][16][KCH];       // 2 x 8 KB
    int wave = threadIdx.x >> 5, lane = threadIdx.x & 31;
    int half = lane >> 4, lr = lane & 15;
    int w = blockIdx.x;                      // B * 128 * 2 blocks
    int b  = w / (128 * 2);
    int rr = w % (128 * 2);
    int mt = rr >> 1, cgrp = rr & 1;
    int m0 = mt * 16;
    int c0 = cgrp * 256 + wave * 64;

    const bf16* aStrip = Pt + ((size_t)b * N_ + m0) * N_;   // 16 x 2048 strip
    const bf16* vb     = v + (size_t)b * CO * N_;

    if (wave == 0) {
        tdm_load_2d_bf16((unsigned)(uintptr_t)&abuf[0][0][0], aStrip + 0,
                         N_, N_, KCH, 16, N_);
        tdm_load_2d_bf16((unsigned)(uintptr_t)&abuf[1][0][0], aStrip + KCH,
                         N_, N_, KCH, 16, N_);
    }

    v8f acc[4];
#pragma unroll
    for (int t = 0; t < 4; ++t) acc[t] = vzero();

    for (int ch = 0; ch < NCH; ++ch) {
        if (wave == 0) {
            if (ch < NCH - 1)
                __builtin_amdgcn_s_wait_tensorcnt(1);
            else
                __builtin_amdgcn_s_wait_tensorcnt(0);
        }
        __syncthreads();                     // chunk ch resident in abuf[ch&1]
        const bf16* arow = &abuf[ch & 1][lr][0];
        int kbase = ch * KCH;
#pragma unroll
        for (int ks = 0; ks < KCH / 32; ++ks) {
            v16bf a = load_a_frag_lds(arow, ks * 32, half);
#pragma unroll
            for (int t = 0; t < 4; ++t) {
                v16bf bb = load_b_frag(vb + (size_t)(c0 + t * 16 + lr) * N_,
                                       kbase + ks * 32, half);
                acc[t] = wmma_bf16(a, bb, acc[t]);
            }
        }
        __syncthreads();                     // all waves done with abuf[ch&1]
        if (wave == 0 && ch + 2 < NCH)
            tdm_load_2d_bf16((unsigned)(uintptr_t)&abuf[ch & 1][0][0],
                             aStrip + (ch + 2) * KCH, N_, N_, KCH, 16, N_);
    }
#pragma unroll
    for (int t = 0; t < 4; ++t)
#pragma unroll
        for (int r = 0; r < 8; ++r)
            out[((size_t)b * N_ + m0 + half * 8 + r) * CO + c0 + t * 16 + lr] = acc[t][r];
}

extern "C" void kernel_launch(void* const* d_in, const int* in_sizes, int n_in,
                              void* d_out, int out_size, void* d_ws, size_t ws_size,
                              hipStream_t stream) {
    const float* x  = (const float*)d_in[0];
    const float* Wq = (const float*)d_in[1];
    const float* Wk = (const float*)d_in[2];
    const float* Wv = (const float*)d_in[3];
    float* out = (float*)d_out;
    char* ws = (char*)d_ws;

    bf16*  Wall = (bf16*)(ws + OFF_WALL);
    bf16*  xT   = (bf16*)(ws + OFF_XT);
    bf16*  qT   = (bf16*)(ws + OFF_QT);
    bf16*  kT   = (bf16*)(ws + OFF_KT);
    bf16*  vv   = (bf16*)(ws + OFF_V);
    float* Mrow = (float*)(ws + OFF_M);
    float* Srow = (float*)(ws + OFF_S);
    bf16*  Pt   = (bf16*)(ws + OFF_PT);

    k_pack_w<<<(CALL * CIN) / 256, 256, 0, stream>>>(Wq, Wk, Wv, Wall);
    k_xpose<<<dim3(N_ / 32, CIN / 32, B_), dim3(32, 8), 0, stream>>>(x, xT);
    k_proj<<<B_ * 64 * 8, 128, 0, stream>>>(Wall, xT, qT, kT, vv);
    k_rowstats<<<(B_ * 128) / 4, 128, 0, stream>>>(kT, qT, Mrow, Srow);
    k_pt<<<(B_ * 128 * 16) / 4, 128, 0, stream>>>(kT, qT, Mrow, Srow, Pt);
    k_out<<<B_ * 128 * 2, 128, 0, stream>>>(Pt, vv, out);
}
